// Encoder_32023276159010
// MI455X (gfx1250) — compile-verified, matrix operand order checked by reading, not run on hardware
//
#include <hip/hip_runtime.h>
#include <hip/hip_bf16.h>

// ---------------------------------------------------------------------------
// GRU encoder for MI455X (gfx1250, wave32, WMMA).
//   B=64, T=512, V=32000, E=300 (pad->320), U=1024, G=3*U=3072
// Strategy:
//   * kernel / recurrent_kernel / gathered embeddings converted to bf16 once,
//     pre-packed into the exact V_WMMA_F32_16X16X32_BF16 per-lane VGPR layout
//     (ISA 7.12.2) -> hot loop is contiguous 32B/lane b128 loads + WMMA.
//   * Hidden state carried BOTH as f32 (exact state update) and row-major
//     bf16 (A-fragments become two b128 loads, zero VALU conversion).
//   * z/r gates fuse xe@Wk and h@Wr into one f32 accumulator (K=320+1024);
//     h gate keeps xh and rh separate (r scales only rh).
//   * ONE persistent kernel runs all 512 timesteps; the sequential dependency
//     is carried by a device-scope atomic grid barrier (256 single-wave
//     blocks are trivially co-resident), eliminating 511 launch latencies.
// ---------------------------------------------------------------------------

#define BB 64
#define TT 512
#define UU 1024
#define EE 300
#define GG 3072
#define NT_TOT 192      // 3072/16 column tiles
#define KS_REC 32       // 1024/32 k-steps (recurrent)
#define KS_IN 10        // 320/32 k-steps (input projection)
#define NBLK 256        // 4 m-tiles x 64 u-tiles

typedef __attribute__((ext_vector_type(16))) __bf16 v16bf;
typedef __attribute__((ext_vector_type(8)))  float  v8f;

union Frag {
    unsigned u[8];
    uint4    q[2];
    v16bf    v;
};

// round-to-nearest-even f32 pair -> packed bf16x2 (low = a, high = b)
__device__ inline unsigned pk2bf(float a, float b) {
    unsigned ua = __float_as_uint(a);
    unsigned ub = __float_as_uint(b);
    ua += 0x7FFFu + ((ua >> 16) & 1u);
    ub += 0x7FFFu + ((ub >> 16) & 1u);
    return (ua >> 16) | (ub & 0xFFFF0000u);
}

__device__ inline unsigned short bf16of(float a) {
    unsigned ua = __float_as_uint(a);
    ua += 0x7FFFu + ((ua >> 16) & 1u);
    return (unsigned short)(ua >> 16);
}

__device__ inline float sigmoidf_(float x) { return 1.0f / (1.0f + __expf(-x)); }

// ---------------------------------------------------------------------------
// Packing kernels (one-shot). Fragment layouts:
//   B-matrix 32x16 (KxN), per lane: col N = 16*nt + (lane&15),
//     K base = 32*ks + (lane<16 ? 0 : 8); dwords 0..3 = K base..base+7 pairs,
//     dwords 4..7 = K base+16..base+23 pairs.
//   A-matrix 16x32 (MxK): same K pattern, row M = (lane&15).
// ---------------------------------------------------------------------------

__global__ __launch_bounds__(32) void pack_rec(const float* __restrict__ rec,
                                               unsigned* __restrict__ frag) {
    const int ks = blockIdx.x;                 // 0..31
    const int nt = blockIdx.y;                 // 0..191
    const int lane = threadIdx.x;
    const int n  = nt * 16 + (lane & 15);
    const int kb = ks * 32 + ((lane < 16) ? 0 : 8);
    unsigned d[8];
#pragma unroll
    for (int j = 0; j < 4; ++j) {
        int k0 = kb + 2 * j;
        int k1 = kb + 16 + 2 * j;
        d[j]     = pk2bf(rec[k0 * GG + n], rec[(k0 + 1) * GG + n]);
        d[4 + j] = pk2bf(rec[k1 * GG + n], rec[(k1 + 1) * GG + n]);
    }
    unsigned* p = frag + (size_t)((nt * KS_REC + ks) * 32 + lane) * 8;
    *(uint4*)(p)     = make_uint4(d[0], d[1], d[2], d[3]);
    *(uint4*)(p + 4) = make_uint4(d[4], d[5], d[6], d[7]);
}

__device__ inline float kread(const float* k, int kk, int n) {
    return (kk < EE) ? k[kk * GG + n] : 0.0f;   // zero-pad K 300..319
}

__global__ __launch_bounds__(32) void pack_ker(const float* __restrict__ ker,
                                               unsigned* __restrict__ frag) {
    const int ks = blockIdx.x;                 // 0..9
    const int nt = blockIdx.y;                 // 0..191
    const int lane = threadIdx.x;
    const int n  = nt * 16 + (lane & 15);
    const int kb = ks * 32 + ((lane < 16) ? 0 : 8);
    unsigned d[8];
#pragma unroll
    for (int j = 0; j < 4; ++j) {
        int k0 = kb + 2 * j;
        int k1 = kb + 16 + 2 * j;
        d[j]     = pk2bf(kread(ker, k0, n), kread(ker, k0 + 1, n));
        d[4 + j] = pk2bf(kread(ker, k1, n), kread(ker, k1 + 1, n));
    }
    unsigned* p = frag + (size_t)((nt * KS_IN + ks) * 32 + lane) * 8;
    *(uint4*)(p)     = make_uint4(d[0], d[1], d[2], d[3]);
    *(uint4*)(p + 4) = make_uint4(d[4], d[5], d[6], d[7]);
}

__device__ inline float eread(const float* e, int kk) {
    return (kk < EE) ? e[kk] : 0.0f;
}

// A-fragments of embedded tokens, keyed (t, batch_tile, kstep): at step t the
// GEMM rows are {b*T + t}, i.e. 16 batches at fixed t.
__global__ __launch_bounds__(32) void pack_xe(const int* __restrict__ x,
                                              const float* __restrict__ emb,
                                              unsigned* __restrict__ frag) {
    const int ks = blockIdx.x;                 // 0..9
    const int mt = blockIdx.y;                 // 0..3
    const int t  = blockIdx.z;                 // 0..511
    const int lane = threadIdx.x;
    const int b  = mt * 16 + (lane & 15);
    const int tok = x[b * TT + t];
    const float* e = emb + (size_t)tok * EE;
    const int kb = ks * 32 + ((lane < 16) ? 0 : 8);
    unsigned d[8];
#pragma unroll
    for (int j = 0; j < 4; ++j) {
        int k0 = kb + 2 * j;
        int k1 = kb + 16 + 2 * j;
        d[j]     = pk2bf(eread(e, k0), eread(e, k0 + 1));
        d[4 + j] = pk2bf(eread(e, k1), eread(e, k1 + 1));
    }
    unsigned* p = frag + (size_t)(((t * 4 + mt) * KS_IN + ks) * 32 + lane) * 8;
    *(uint4*)(p)     = make_uint4(d[0], d[1], d[2], d[3]);
    *(uint4*)(p + 4) = make_uint4(d[4], d[5], d[6], d[7]);
}

// input hidden f32 -> row-major bf16, and zero the grid-sync counter
__global__ __launch_bounds__(256) void init_state(const float* __restrict__ h,
                                                  unsigned short* __restrict__ hb,
                                                  unsigned* __restrict__ sync) {
    int i = blockIdx.x * 256 + threadIdx.x;    // 0 .. 65535
    hb[i] = bf16of(h[i]);
    if (i == 0) *sync = 0u;
}

// ---------------------------------------------------------------------------
// Persistent GRU: 256 blocks x 1 wave; block bid -> (mt = bid&3, ut = bid>>2).
// All 512 timesteps in one launch; device-scope atomic barrier between steps.
// Per step per wave: 126 v_wmma_f32_16x16x32_bf16.
// ---------------------------------------------------------------------------
__global__ __launch_bounds__(32) void gru_persist(
        const float* __restrict__ hidden,
        float* __restrict__ hbuf0, float* __restrict__ hbuf1,
        unsigned short* __restrict__ hbf0, unsigned short* __restrict__ hbf1,
        const unsigned* __restrict__ recFrag,
        const unsigned* __restrict__ kerFrag,
        const unsigned* __restrict__ xeFrag,
        const float* __restrict__ bias,
        float* __restrict__ out,
        unsigned* __restrict__ sync) {
    const int lane = threadIdx.x;
    const int mt = blockIdx.x & 3;             // batch tile 0..3
    const int ut = blockIdx.x >> 2;            // u tile 0..63
    const int row  = lane & 15;
    const int koff = (lane < 16) ? 0 : 8;

    const int ntz = ut, ntr = 64 + ut, nth = 128 + ut;
    const unsigned* pz0 = recFrag + (size_t)(ntz * KS_REC * 32 + lane) * 8;
    const unsigned* pr0 = recFrag + (size_t)(ntr * KS_REC * 32 + lane) * 8;
    const unsigned* ph0 = recFrag + (size_t)(nth * KS_REC * 32 + lane) * 8;
    const unsigned* qz0 = kerFrag + (size_t)(ntz * KS_IN * 32 + lane) * 8;
    const unsigned* qr0 = kerFrag + (size_t)(ntr * KS_IN * 32 + lane) * 8;
    const unsigned* qh0 = kerFrag + (size_t)(nth * KS_IN * 32 + lane) * 8;

    const int u = ut * 16 + row;
    const float b0z = bias[u],            b1z = bias[GG + u];
    const float b0r = bias[UU + u],       b1r = bias[GG + UU + u];
    const float b0h = bias[2 * UU + u],   b1h = bias[GG + 2 * UU + u];

    for (int t = 0; t < TT; ++t) {
        const float*          hsrc    = (t == 0) ? hidden : ((t & 1) ? hbuf0 : hbuf1);
        const unsigned short* hsrc_bf = (t == 0) ? hbf1   : ((t & 1) ? hbf0  : hbf1);
        float*          hdst    = (t & 1) ? hbuf1 : hbuf0;
        unsigned short* hdst_bf = (t & 1) ? hbf1  : hbf0;

        v8f cz = {}, cr = {}, chx = {}, chr = {};

        const unsigned short* hrow = hsrc_bf + (mt * 16 + row) * UU + koff;
        const unsigned* pz = pz0;
        const unsigned* pr = pr0;
        const unsigned* ph = ph0;

        // ---- recurrent part: K = 1024 ----
        for (int ks = 0; ks < KS_REC; ++ks) {
            Frag fa, fz, fr, fh;
            fa.q[0] = *(const uint4*)(hrow);        // K kb..kb+7   (8 bf16)
            fa.q[1] = *(const uint4*)(hrow + 16);   // K kb+16..kb+23
            hrow += 32;
            fz.q[0] = *(const uint4*)(pz); fz.q[1] = *(const uint4*)(pz + 4);
            fr.q[0] = *(const uint4*)(pr); fr.q[1] = *(const uint4*)(pr + 4);
            fh.q[0] = *(const uint4*)(ph); fh.q[1] = *(const uint4*)(ph + 4);
            pz += 256; pr += 256; ph += 256;

            cz  = __builtin_amdgcn_wmma_f32_16x16x32_bf16(false, fa.v, false, fz.v,
                                                          (short)0, cz, false, false);
            cr  = __builtin_amdgcn_wmma_f32_16x16x32_bf16(false, fa.v, false, fr.v,
                                                          (short)0, cr, false, false);
            chr = __builtin_amdgcn_wmma_f32_16x16x32_bf16(false, fa.v, false, fh.v,
                                                          (short)0, chr, false, false);
        }

        // ---- input projection part: K = 320 (pre-packed xe fragments) ----
        const unsigned* pa = xeFrag + (size_t)((t * 4 + mt) * KS_IN * 32 + lane) * 8;
        const unsigned* qz = qz0;
        const unsigned* qr = qr0;
        const unsigned* qh = qh0;
        for (int ks = 0; ks < KS_IN; ++ks) {
            Frag fa, fz, fr, fh;
            fa.q[0] = *(const uint4*)(pa); fa.q[1] = *(const uint4*)(pa + 4);
            fz.q[0] = *(const uint4*)(qz); fz.q[1] = *(const uint4*)(qz + 4);
            fr.q[0] = *(const uint4*)(qr); fr.q[1] = *(const uint4*)(qr + 4);
            fh.q[0] = *(const uint4*)(qh); fh.q[1] = *(const uint4*)(qh + 4);
            pa += 256; qz += 256; qr += 256; qh += 256;

            cz  = __builtin_amdgcn_wmma_f32_16x16x32_bf16(false, fa.v, false, fz.v,
                                                          (short)0, cz, false, false);
            cr  = __builtin_amdgcn_wmma_f32_16x16x32_bf16(false, fa.v, false, fr.v,
                                                          (short)0, cr, false, false);
            chx = __builtin_amdgcn_wmma_f32_16x16x32_bf16(false, fa.v, false, fh.v,
                                                          (short)0, chx, false, false);
        }

        // ---- gate epilogue ----
        // C layout: VGPR i -> M = i (lanes 0..15) / i+8 (16..31), N = lane&15
#pragma unroll
        for (int i = 0; i < 8; ++i) {
            const int b = mt * 16 + i + koff;
            float z  = sigmoidf_(cz[i] + b0z + b1z);
            float r  = sigmoidf_(cr[i] + b0r + b1r);
            float hh = tanhf((chx[i] + b0h) + r * (chr[i] + b1h));
            float ho = hsrc[b * UU + u];
            float hn = z * ho + (1.0f - z) * hh;
            hdst[b * UU + u]    = hn;
            hdst_bf[b * UU + u] = bf16of(hn);
            out[(size_t)(b * TT + t) * UU + u] = hn;
        }

        // ---- grid barrier: release our writes, wait for all 256 blocks ----
        __threadfence();
        if (lane == 0) {
            __hip_atomic_fetch_add(sync, 1u, __ATOMIC_ACQ_REL,
                                   __HIP_MEMORY_SCOPE_AGENT);
            const unsigned tgt = (unsigned)NBLK * (unsigned)(t + 1);
            while (__hip_atomic_load(sync, __ATOMIC_ACQUIRE,
                                     __HIP_MEMORY_SCOPE_AGENT) < tgt) {
                __builtin_amdgcn_s_sleep(2);
            }
        }
        __syncthreads();      // single-wave WG: lanes reconverge with lane 0
        __threadfence();
    }
}

// ---------------------------------------------------------------------------
extern "C" void kernel_launch(void* const* d_in, const int* in_sizes, int n_in,
                              void* d_out, int out_size, void* d_ws, size_t ws_size,
                              hipStream_t stream) {
    const int*   x      = (const int*)d_in[0];
    const float* hidden = (const float*)d_in[1];
    const float* emb    = (const float*)d_in[2];
    const float* ker    = (const float*)d_in[3];
    const float* rec    = (const float*)d_in[4];
    const float* bias   = (const float*)d_in[5];
    float* out = (float*)d_out;

    // workspace layout (dwords): ~30 MB total
    unsigned* recFrag = (unsigned*)d_ws;                       // 1,572,864 dw
    unsigned* kerFrag = recFrag + (size_t)NT_TOT * KS_REC * 32 * 8;
    unsigned* xeFrag  = kerFrag + (size_t)NT_TOT * KS_IN * 32 * 8;     // + 491,520 dw
    float* hbuf0 = (float*)(xeFrag + (size_t)TT * 4 * KS_IN * 32 * 8); // + 5,242,880 dw
    float* hbuf1 = hbuf0 + BB * UU;
    unsigned short* hbf0 = (unsigned short*)(hbuf1 + BB * UU);
    unsigned short* hbf1 = hbf0 + BB * UU;
    unsigned* syncc = (unsigned*)(hbf1 + BB * UU);

    pack_rec<<<dim3(KS_REC, NT_TOT), 32, 0, stream>>>(rec, recFrag);
    pack_ker<<<dim3(KS_IN,  NT_TOT), 32, 0, stream>>>(ker, kerFrag);
    pack_xe <<<dim3(KS_IN, 4, TT),   32, 0, stream>>>(x, emb, xeFrag);
    // t=0 reads bf16 h from hbf1; hbf1 is first overwritten at t=1 (safe).
    // Also zeroes the grid-sync counter (required fresh every call).
    init_state<<<BB * UU / 256, 256, 0, stream>>>(hidden, hbf1, syncc);

    gru_persist<<<NBLK, 32, 0, stream>>>(hidden, hbuf0, hbuf1, hbf0, hbf1,
                                         recFrag, kerFrag, xeFrag, bias, out,
                                         syncc);

    // final state (t=511 wrote hbuf1) -> d_out tail
    hipMemcpyAsync(out + (size_t)BB * TT * UU, hbuf1,
                   (size_t)BB * UU * sizeof(float),
                   hipMemcpyDeviceToDevice, stream);
}
